// MixedAttention_48567490183355
// MI455X (gfx1250) — compile-verified, matrix operand order checked by reading, not run on hardware
//
#include <hip/hip_runtime.h>
#include <hip/hip_bf16.h>

// ---------- types ----------
typedef __attribute__((ext_vector_type(16))) _Float16 v16h;
typedef __attribute__((ext_vector_type(8)))  _Float16 v8h;
typedef __attribute__((ext_vector_type(8)))  float    v8f;

#define DIM 512
#define TQ  192
#define BB  4
#define AH  8
#define CH  4
#define DH  64
#define BT  (BB*TQ)   // 768

__device__ __forceinline__ v8f vzero8() {
  v8f z;
#pragma unroll
  for (int i = 0; i < 8; i++) z[i] = 0.f;
  return z;
}

__device__ __forceinline__ v8f wmma32(v16h a, v16h b, v8f c) {
  return __builtin_amdgcn_wmma_f32_16x16x32_f16(false, a, false, b, (short)0, c,
                                                false, false);
}

// A/B fragment loader for v_wmma_f32_16x16x32_f16.
// ISA 16-bit 16x32 layout: lane L holds row (L%16); halves i=0..7 -> k=kb+i,
// i=8..15 -> k=kb+16+(i-8), kb = 8*(L>=16).  => two contiguous b128 loads.
__device__ __forceinline__ v16h load_frag(const _Float16* rowp, int k0, int lane) {
  const _Float16* p = rowp + k0 + ((lane >> 4) << 3);
  v8h lo = *(const v8h*)(p);
  v8h hi = *(const v8h*)(p + 16);
  v16h r;
#pragma unroll
  for (int i = 0; i < 8; i++) { r[i] = lo[i]; r[i + 8] = hi[i]; }
  return r;
}

__device__ __forceinline__ float half16_max(float v) {
#pragma unroll
  for (int m = 1; m < 16; m <<= 1) v = fmaxf(v, __shfl_xor(v, m, 32));
  return v;
}
__device__ __forceinline__ float half16_sum(float v) {
#pragma unroll
  for (int m = 1; m < 16; m <<= 1) v += __shfl_xor(v, m, 32);
  return v;
}

// ---------- LayerNorm (both branches), f32 -> f16 ----------
__global__ __launch_bounds__(256) void ln_kernel(
    const float* __restrict__ x,
    const float* __restrict__ g1, const float* __restrict__ b1,
    const float* __restrict__ g2, const float* __restrict__ b2,
    _Float16* __restrict__ xn1, _Float16* __restrict__ xn2) {
  int row = blockIdx.x, tid = threadIdx.x;
  __shared__ float red[256];
  const float* xr = x + (size_t)row * DIM;
  float a = xr[tid], b = xr[tid + 256];
  red[tid] = a + b;
  __syncthreads();
  for (int off = 128; off > 0; off >>= 1) {
    if (tid < off) red[tid] += red[tid + off];
    __syncthreads();
  }
  float mu = red[0] * (1.f / DIM);
  __syncthreads();
  float da = a - mu, db = b - mu;
  red[tid] = da * da + db * db;
  __syncthreads();
  for (int off = 128; off > 0; off >>= 1) {
    if (tid < off) red[tid] += red[tid + off];
    __syncthreads();
  }
  float rs = rsqrtf(red[0] * (1.f / DIM) + 1e-5f);
  size_t o = (size_t)row * DIM;
  xn1[o + tid]       = (_Float16)(da * rs * g1[tid]       + b1[tid]);
  xn1[o + tid + 256] = (_Float16)(db * rs * g1[tid + 256] + b1[tid + 256]);
  xn2[o + tid]       = (_Float16)(da * rs * g2[tid]       + b2[tid]);
  xn2[o + tid + 256] = (_Float16)(db * rs * g2[tid + 256] + b2[tid + 256]);
}

// ---------- weight transpose: W[K,N] f32 -> WT[N,K] f16 ----------
__global__ __launch_bounds__(256) void wt_kernel(const float* __restrict__ W,
                                                 _Float16* __restrict__ WT,
                                                 int K, int N) {
  int idx = blockIdx.x * 256 + threadIdx.x;
  if (idx >= K * N) return;
  int n = idx / K, k = idx - n * K;
  WT[(size_t)n * K + k] = (_Float16)W[(size_t)k * N + n];
}

// ---------- per-head transpose: src[BT,src_ld] (col_off + h*64 + d) ->
//            dst[(b*H+h)*64 + d][192] ----------
__global__ __launch_bounds__(256) void headT_kernel(const _Float16* __restrict__ src,
                                                    _Float16* __restrict__ dst,
                                                    int src_ld, int col_off, int H) {
  int idx = blockIdx.x * 256 + threadIdx.x;
  int total = BB * H * DH * TQ;
  if (idx >= total) return;
  int t  = idx % TQ;
  int d  = (idx / TQ) % DH;
  int bh = idx / (TQ * DH);
  int b  = bh / H, h = bh % H;
  dst[idx] = src[(size_t)(b * TQ + t) * src_ld + col_off + h * DH + d];
}

// ---------- generic f16 WMMA GEMM: C[M,N] = A[M,K] @ WT[N,K]^T (+bias) ----------
__global__ __launch_bounds__(128) void gemm16(const _Float16* __restrict__ A, int lda,
                                              const _Float16* __restrict__ WT, int ldw,
                                              int K,
                                              const float* __restrict__ bias,
                                              float* __restrict__ Cf,
                                              _Float16* __restrict__ Ch,
                                              int ldc, int accumulate) {
  int lane = threadIdx.x & 31, wave = threadIdx.x >> 5;
  int mt = blockIdx.x, nt = blockIdx.y * 4 + wave;
  const _Float16* arow = A  + (size_t)(mt * 16 + (lane & 15)) * lda;
  const _Float16* brow = WT + (size_t)(nt * 16 + (lane & 15)) * ldw;
  v8f acc = vzero8();
  for (int k0 = 0; k0 < K; k0 += 32) {
    __builtin_prefetch(brow + k0 + 128, 0, 1);
    v16h af = load_frag(arow, k0, lane);
    v16h bf = load_frag(brow, k0, lane);
    acc = wmma32(af, bf, acc);
  }
  int col = nt * 16 + (lane & 15);
  float bv = bias ? bias[col] : 0.f;
#pragma unroll
  for (int r = 0; r < 8; r++) {
    int row = mt * 16 + r + 8 * (lane >> 4);
    float v = acc[r] + bv;
    if (Ch) Ch[(size_t)row * ldc + col] = (_Float16)v;
    else if (accumulate) Cf[(size_t)row * ldc + col] += v;
    else Cf[(size_t)row * ldc + col] = v;
  }
}

// ---------- attention: 4 waves/block, one wave per q-tile; K staged in LDS ----------
__global__ __launch_bounds__(128) void attn_kernel(const _Float16* __restrict__ qkv,
                                                   const _Float16* __restrict__ vT,
                                                   _Float16* __restrict__ attn_out) {
  int tid = threadIdx.x;
  int lane = tid & 31, wave = tid >> 5;
  int g = blockIdx.x;                  // BB*AH*3
  int qg = g % 3, h = (g / 3) % AH, b = g / (3 * AH);
  int qt = qg * 4 + wave;

  __shared__ _Float16 kst[TQ * DH];    // 24 KB : K head-slice
  __shared__ _Float16 P[4 * 16 * TQ];  // 24 KB : per-wave probabilities

  // stage K[t][d] for this (b,h)
  for (int i = tid; i < TQ * 8; i += 128) {
    int t = i >> 3, c = (i & 7) << 3;
    *(v8h*)&kst[t * DH + c] =
        *(const v8h*)&qkv[(size_t)(b * TQ + t) * (3 * DIM) + DIM + h * DH + c];
  }
  __syncthreads();

  const _Float16* qbase =
      qkv + (size_t)(b * TQ + qt * 16 + (lane & 15)) * (3 * DIM) + h * DH;
  v16h qf0 = load_frag(qbase, 0, lane);
  v16h qf1 = load_frag(qbase, 32, lane);

  v8f acc[12];
#pragma unroll
  for (int jt = 0; jt < 12; jt++) {
    const _Float16* kbase = &kst[(jt * 16 + (lane & 15)) * DH];
    v8f a = vzero8();
    a = wmma32(qf0, load_frag(kbase, 0, lane), a);
    a = wmma32(qf1, load_frag(kbase, 32, lane), a);
    acc[jt] = a;
  }
  const float scale = 0.125f;  // 64^-0.5
  float m[8], l[8];
#pragma unroll
  for (int r = 0; r < 8; r++) m[r] = -3.4e38f;
#pragma unroll
  for (int jt = 0; jt < 12; jt++)
#pragma unroll
    for (int r = 0; r < 8; r++) m[r] = fmaxf(m[r], acc[jt][r]);
#pragma unroll
  for (int r = 0; r < 8; r++) { m[r] = half16_max(m[r]) * scale; l[r] = 0.f; }
#pragma unroll
  for (int jt = 0; jt < 12; jt++)
#pragma unroll
    for (int r = 0; r < 8; r++) {
      float p = __expf(fmaf(acc[jt][r], scale, -m[r]));
      acc[jt][r] = p;
      l[r] += p;
    }
#pragma unroll
  for (int r = 0; r < 8; r++) l[r] = half16_sum(l[r]);

  _Float16* Pw = &P[wave * 16 * TQ];
#pragma unroll
  for (int jt = 0; jt < 12; jt++)
#pragma unroll
    for (int r = 0; r < 8; r++)
      Pw[(r + 8 * (lane >> 4)) * TQ + jt * 16 + (lane & 15)] = (_Float16)acc[jt][r];
  __syncthreads();

  const _Float16* vall = vT + (size_t)((b * AH + h) * DH) * TQ;
#pragma unroll
  for (int dt = 0; dt < 4; dt++) {
    v8f o = vzero8();
    const _Float16* vrow = vall + (size_t)(dt * 16 + (lane & 15)) * TQ;
    const _Float16* prow = &Pw[(lane & 15) * TQ];
#pragma unroll
    for (int kk = 0; kk < 6; kk++)
      o = wmma32(load_frag(prow, kk * 32, lane), load_frag(vrow, kk * 32, lane), o);
#pragma unroll
    for (int r = 0; r < 8; r++) {
      int row = b * TQ + qt * 16 + r + 8 * (lane >> 4);
      int col = h * DH + dt * 16 + (lane & 15);
      attn_out[(size_t)row * DIM + col] = (_Float16)(o[r] / l[r]);
    }
  }
}

// ---------- trittention: 4 waves/block, one wave per q-tile ----------
// score = (c_q .* a_s) @ b_t^T  : the a-product folds into the A operand once
// per s and is amortized over all 12 t-tiles.
__global__ __launch_bounds__(128) void trit_kernel(const _Float16* __restrict__ proj,
                                                   const _Float16* __restrict__ dT,
                                                   const _Float16* __restrict__ eT,
                                                   _Float16* __restrict__ zout) {
  int tid = threadIdx.x;
  int lane = tid & 31, wave = tid >> 5;
  int g = blockIdx.x;                  // BB*CH*3
  int qg = g % 3, h = (g / 3) % CH, b = g / (3 * CH);
  int qt = qg * 4 + wave;
  const int LD = 5 * CH * DH;          // 1280
  const float inv64 = 1.f / 64.f;

  __shared__ _Float16 bst[TQ * DH];        // 24 KB: b head-slice; margT aliases later
  __shared__ _Float16 margSh[4 * 16 * TQ]; // 24 KB: per-wave margS (f16)

  // stage b_[t][d] for this (b,h)
  for (int i = tid; i < TQ * 8; i += 128) {
    int t = i >> 3, c = (i & 7) << 3;
    *(v8h*)&bst[t * DH + c] =
        *(const v8h*)&proj[(size_t)(b * TQ + t) * LD + 256 + h * DH + c];
  }
  __syncthreads();

  const _Float16* cbase =
      proj + (size_t)(b * TQ + qt * 16 + (lane & 15)) * LD + 512 + h * DH;
  v16h cf0 = load_frag(cbase, 0, lane);
  v16h cf1 = load_frag(cbase, 32, lane);

  // ---- pass 1: raw row max over all (s,t) (scale deferred) ----
  float m[8];
#pragma unroll
  for (int r = 0; r < 8; r++) m[r] = -3.4e38f;
  for (int s = 0; s < TQ; s++) {
    const _Float16* abase = proj + (size_t)(b * TQ + s) * LD + h * DH;
    v16h a0 = cf0 * load_frag(abase, 0, lane);   // (c .* a_s): once per s
    v16h a1 = cf1 * load_frag(abase, 32, lane);
#pragma unroll
    for (int tt = 0; tt < 12; tt++) {
      const _Float16* brow = &bst[(tt * 16 + (lane & 15)) * DH];
      v8f sc = vzero8();
      sc = wmma32(a0, load_frag(brow, 0, lane), sc);
      sc = wmma32(a1, load_frag(brow, 32, lane), sc);
#pragma unroll
      for (int r = 0; r < 8; r++) m[r] = fmaxf(m[r], sc[r]);
    }
  }
#pragma unroll
  for (int r = 0; r < 8; r++) m[r] = half16_max(m[r]) * inv64;

  // ---- pass 2: exponentiate, accumulate marginals ----
  v8f mT[12];
#pragma unroll
  for (int tt = 0; tt < 12; tt++) mT[tt] = vzero8();
  float lacc[8];
#pragma unroll
  for (int r = 0; r < 8; r++) lacc[r] = 0.f;
  _Float16* mSw = &margSh[wave * 16 * TQ];

  for (int s = 0; s < TQ; s++) {
    const _Float16* abase = proj + (size_t)(b * TQ + s) * LD + h * DH;
    v16h a0 = cf0 * load_frag(abase, 0, lane);
    v16h a1 = cf1 * load_frag(abase, 32, lane);
    float rs[8];
#pragma unroll
    for (int r = 0; r < 8; r++) rs[r] = 0.f;
#pragma unroll
    for (int tt = 0; tt < 12; tt++) {
      const _Float16* brow = &bst[(tt * 16 + (lane & 15)) * DH];
      v8f sc = vzero8();
      sc = wmma32(a0, load_frag(brow, 0, lane), sc);
      sc = wmma32(a1, load_frag(brow, 32, lane), sc);
#pragma unroll
      for (int r = 0; r < 8; r++) {
        float p = __expf(fmaf(sc[r], inv64, -m[r]));
        mT[tt][r] += p;
        rs[r] += p;
      }
    }
#pragma unroll
    for (int r = 0; r < 8; r++) { rs[r] = half16_sum(rs[r]); lacc[r] += rs[r]; }
    if ((lane & 15) == 0) {
#pragma unroll
      for (int r = 0; r < 8; r++)
        mSw[(r + 8 * (lane >> 4)) * TQ + s] = (_Float16)rs[r];
    }
  }
  __syncthreads();          // all waves done reading bst -> safe to alias

  _Float16* mTw = &bst[0] + wave * 16 * TQ;  // margT tiles alias the b-stage
#pragma unroll
  for (int tt = 0; tt < 12; tt++)
#pragma unroll
    for (int r = 0; r < 8; r++)
      mTw[(r + 8 * (lane >> 4)) * TQ + tt * 16 + (lane & 15)] = (_Float16)mT[tt][r];
  __syncthreads();

  float linv[8];
#pragma unroll
  for (int r = 0; r < 8; r++) linv[r] = 1.f / lacc[r];

  const _Float16* dall = dT + (size_t)((b * CH + h) * DH) * TQ;
  const _Float16* eall = eT + (size_t)((b * CH + h) * DH) * TQ;
#pragma unroll
  for (int dt = 0; dt < 4; dt++) {
    v8f z = vzero8();
    const _Float16* drow = dall + (size_t)(dt * 16 + (lane & 15)) * TQ;
    const _Float16* erow = eall + (size_t)(dt * 16 + (lane & 15)) * TQ;
    const _Float16* srow = &mSw[(lane & 15) * TQ];
    const _Float16* trow = &mTw[(lane & 15) * TQ];
#pragma unroll
    for (int kk = 0; kk < 6; kk++) {
      z = wmma32(load_frag(srow, kk * 32, lane), load_frag(drow, kk * 32, lane), z);
      z = wmma32(load_frag(trow, kk * 32, lane), load_frag(erow, kk * 32, lane), z);
    }
#pragma unroll
    for (int r = 0; r < 8; r++) {
      int row = b * TQ + qt * 16 + r + 8 * (lane >> 4);
      int col = h * DH + dt * 16 + (lane & 15);
      zout[(size_t)row * (CH * DH) + col] = (_Float16)(z[r] * linv[r]);
    }
  }
}

// ---------- host ----------
extern "C" void kernel_launch(void* const* d_in, const int* in_sizes, int n_in,
                              void* d_out, int out_size, void* d_ws, size_t ws_size,
                              hipStream_t stream) {
  const float* x      = (const float*)d_in[0];
  const float* ln1_g  = (const float*)d_in[1];
  const float* ln1_b  = (const float*)d_in[2];
  const float* Wqkv   = (const float*)d_in[3];
  const float* Wo     = (const float*)d_in[4];
  const float* bo     = (const float*)d_in[5];
  const float* ln2_g  = (const float*)d_in[6];
  const float* ln2_b  = (const float*)d_in[7];
  const float* Wabcde = (const float*)d_in[8];
  const float* babcde = (const float*)d_in[9];
  const float* Wp     = (const float*)d_in[10];
  const float* bp     = (const float*)d_in[11];
  float* out = (float*)d_out;

  char* w = (char*)d_ws;
  auto alloc = [&](size_t elems) {
    void* p = (void*)w;
    w += (elems * sizeof(_Float16) + 255) & ~(size_t)255;
    return (_Float16*)p;
  };
  _Float16* xn1      = alloc((size_t)BT * DIM);
  _Float16* xn2      = alloc((size_t)BT * DIM);
  _Float16* WqkvT    = alloc((size_t)1536 * 512);
  _Float16* WoT      = alloc((size_t)512 * 512);
  _Float16* WabcdeT  = alloc((size_t)1280 * 512);
  _Float16* WpT      = alloc((size_t)512 * 256);
  _Float16* qkv      = alloc((size_t)BT * 1536);
  _Float16* vT       = alloc((size_t)BB * AH * DH * TQ);
  _Float16* attn_out = alloc((size_t)BT * DIM);
  _Float16* proj     = alloc((size_t)BT * 1280);
  _Float16* dTr      = alloc((size_t)BB * CH * DH * TQ);
  _Float16* eTr      = alloc((size_t)BB * CH * DH * TQ);
  _Float16* zout     = alloc((size_t)BT * 256);

  ln_kernel<<<BT, 256, 0, stream>>>(x, ln1_g, ln1_b, ln2_g, ln2_b, xn1, xn2);

  wt_kernel<<<(512 * 1536 + 255) / 256, 256, 0, stream>>>(Wqkv, WqkvT, 512, 1536);
  wt_kernel<<<(512 * 512 + 255) / 256, 256, 0, stream>>>(Wo, WoT, 512, 512);
  wt_kernel<<<(512 * 1280 + 255) / 256, 256, 0, stream>>>(Wabcde, WabcdeT, 512, 1280);
  wt_kernel<<<(256 * 512 + 255) / 256, 256, 0, stream>>>(Wp, WpT, 256, 512);

  // qkv = xn1 @ Wqkv      [768,1536]
  gemm16<<<dim3(BT / 16, 1536 / 64), 128, 0, stream>>>(xn1, DIM, WqkvT, DIM, DIM,
                                                       nullptr, nullptr, qkv, 1536, 0);
  headT_kernel<<<(BB * AH * DH * TQ + 255) / 256, 256, 0, stream>>>(qkv, vT, 1536,
                                                                    1024, AH);
  attn_kernel<<<BB * AH * 3, 128, 0, stream>>>(qkv, vT, attn_out);
  // out = attn_out @ Wo + bo
  gemm16<<<dim3(BT / 16, 512 / 64), 128, 0, stream>>>(attn_out, DIM, WoT, DIM, DIM,
                                                      bo, out, nullptr, DIM, 0);
  // proj = xn2 @ Wabcde + babcde   [768,1280]
  gemm16<<<dim3(BT / 16, 1280 / 64), 128, 0, stream>>>(xn2, DIM, WabcdeT, DIM, DIM,
                                                       babcde, nullptr, proj, 1280, 0);
  headT_kernel<<<(BB * CH * DH * TQ + 255) / 256, 256, 0, stream>>>(proj, dTr, 1280,
                                                                    768, CH);
  headT_kernel<<<(BB * CH * DH * TQ + 255) / 256, 256, 0, stream>>>(proj, eTr, 1280,
                                                                    1024, CH);
  trit_kernel<<<BB * CH * 3, 128, 0, stream>>>(proj, dTr, eTr, zout);
  // out += zout @ Wp + bp
  gemm16<<<dim3(BT / 16, 512 / 64), 128, 0, stream>>>(zout, 256, WpT, 256, 256,
                                                      bp, out, nullptr, DIM, 1);
}
